// Net_7645041787060
// MI455X (gfx1250) — matrix operand order repaired and checked by reading, not compiled, not run on hardware
//
#include <hip/hip_runtime.h>

// ---------------------------------------------------------------------------
// Point Transformer forward for MI455X (gfx1250, wave32, WMMA + TDM).
//  * C x C linears -> v_wmma_f32_16x16x32_f16, fp32 accumulate.
//      - A tile: float4 global loads (phase-split), packed half2 ds_store_b64
//      - B tile: Tensor Data Mover (tensor_load_to_lds) DMA of the f32
//        weight tile, s_wait_tensorcnt fence, cooperative f16 pack
//      - fragments: two ds_load_b128 per matrix (ISA lane layout runs)
//  * KNN Gram matrix -> v_wmma_f32_16x16x4_f32 (K=3 padded to 4), fused
//    top-16 register insertion sort over LDS distance chunks.
//  * BN statistics via LDS tree reduction, softmax+weighted-sum fused.
// ---------------------------------------------------------------------------

typedef __attribute__((ext_vector_type(16))) _Float16 v16h;
typedef __attribute__((ext_vector_type(8)))  _Float16 v8h;
typedef __attribute__((ext_vector_type(8)))  float    v8f;
typedef __attribute__((ext_vector_type(2)))  float    v2f;
typedef __attribute__((ext_vector_type(4)))  unsigned int v4u;
typedef __attribute__((ext_vector_type(8)))  int      v8i;
typedef __attribute__((ext_vector_type(4)))  int      v4i;

#define BN_EPS 1e-5f
#define BATCH 8

static inline int ceildiv(int a, int b) { return (a + b - 1) / b; }

__device__ __forceinline__ unsigned int pack_h2(float a, float b) {
  union { _Float16 h[2]; unsigned int u; } p;
  p.h[0] = (_Float16)a; p.h[1] = (_Float16)b;
  return p.u;
}

// ---------------------------------------------------------------------------
// WMMA GEMM: Y[M,N] = X[M,K] @ W[K,N] + bias (+residual) (opt relu)
// Block = 128 threads (4 waves); block computes a 64x16 output tile.
// ---------------------------------------------------------------------------
__global__ void linear_wmma_kernel(const float* __restrict__ X,
                                   const float* __restrict__ W,
                                   const float* __restrict__ bias,
                                   const float* __restrict__ residual,
                                   float* __restrict__ Y,
                                   int M, int K, int N, int relu) {
  __shared__ unsigned int As32[64][16];   // 64 rows x 32 halves (packed)
  __shared__ float        Bsf[32][16];    // TDM destination: f32 tile [k][n]
  __shared__ unsigned int Bst32[16][16];  // packed halves, [n][k/2]

  const int tid   = threadIdx.x;
  const int wave  = tid >> 5;
  const int lane  = tid & 31;
  const int half_ = lane >> 4;
  const int lrow  = lane & 15;
  const int n0    = blockIdx.x * 16;
  const int m0    = blockIdx.y * 64;
  const bool fullM = (m0 + 64) <= M;
  // A-tile staging coordinates (float4 granularity): 64 rows x 8 float4
  const int ar0 = tid >> 3;       // rows ar0, ar0+16, ar0+32, ar0+48
  const int ac4 = tid & 7;        // float4 column

  v8f acc = {};
  for (int k0 = 0; k0 < K; k0 += 32) {
    // ---- stage A: phase-split loads, then pack to LDS ----
    float4 fa[4];
    if (fullM) {
#pragma unroll
      for (int i = 0; i < 4; ++i) {
        int r = ar0 + 16 * i;
        fa[i] = *(const float4*)(X + (size_t)(m0 + r) * K + k0 + 4 * ac4);
      }
      if (k0 + 32 < K)
        __builtin_prefetch(X + (size_t)(m0 + ar0) * K + k0 + 32 + 4 * ac4, 0, 1);
    } else {
#pragma unroll
      for (int i = 0; i < 4; ++i) {
        int gm = m0 + ar0 + 16 * i;
        float4 z = {0.f, 0.f, 0.f, 0.f};
        fa[i] = (gm < M) ? *(const float4*)(X + (size_t)gm * K + k0 + 4 * ac4) : z;
      }
    }
#pragma unroll
    for (int i = 0; i < 4; ++i) {
      int r = ar0 + 16 * i;
      uint2 u;
      u.x = pack_h2(fa[i].x, fa[i].y);
      u.y = pack_h2(fa[i].z, fa[i].w);
      *(uint2*)&As32[r][2 * ac4] = u;
    }

    // ---- B tile via Tensor Data Mover: 2D tile 16(n) x 32(k), f32 ----
    if (wave == 0) {
      unsigned long long ga =
          (unsigned long long)(size_t)(W + (size_t)k0 * N + n0);
      unsigned int ldsAddr = (unsigned int)(size_t)(&Bsf[0][0]);
      // D# group0: count=1, lds_addr, 57b global addr, type=2
      v4u g0 = { 1u, ldsAddr,
                 (unsigned int)(ga & 0xFFFFFFFFull),
                 (unsigned int)((ga >> 32) & 0x01FFFFFFull) | (2u << 30) };
      // D# group1: data_size=4B(code 2); tensor_dim0=N, tensor_dim1=K;
      // tile_dim0=16, tile_dim1=32; tensor_dim0_stride=N
      unsigned int uN = (unsigned int)N, uK = (unsigned int)K;
      v8i g1 = { (int)(2u << 16),
                 (int)((uN & 0xFFFFu) << 16),
                 (int)((uN >> 16) | ((uK & 0xFFFFu) << 16)),
                 (int)((uK >> 16) | (16u << 16)),
                 (int)32u,
                 (int)uN,
                 0, 0 };
      v4i z4 = { 0, 0, 0, 0 };
      v8i z8 = { 0, 0, 0, 0, 0, 0, 0, 0 };
      __builtin_amdgcn_tensor_load_to_lds(g0, g1, z4, z4, z8, 0);
      __builtin_amdgcn_s_wait_tensorcnt(0);
    }
    __syncthreads();

    // ---- pack B tile to halves, transposed to [n][k] for fast frags ----
#pragma unroll
    for (int i = 0; i < 2; ++i) {
      int e = tid + 128 * i;
      int nn = e & 15, k2 = e >> 4;
      Bst32[nn][k2] = pack_h2(Bsf[2 * k2][nn], Bsf[2 * k2 + 1][nn]);
    }
    __syncthreads();

    // ---- fragments: contiguous 8-half runs -> ds_load_b128 pairs ----
    const int mrow = 16 * wave + lrow;
    v8h alo = *(const v8h*)&As32[mrow][half_ ? 4 : 0];        // k: h*8 ..
    v8h ahi = *(const v8h*)&As32[mrow][8 + (half_ ? 4 : 0)];  // k: 16+h*8 ..
    v8h blo = *(const v8h*)&Bst32[lrow][half_ ? 4 : 0];        // k: h*8 ..
    v8h bhi = *(const v8h*)&Bst32[lrow][8 + (half_ ? 4 : 0)];  // k: 16+h*8 ..
    v16h a = __builtin_shufflevector(alo, ahi, 0, 1, 2, 3, 4, 5, 6, 7,
                                     8, 9, 10, 11, 12, 13, 14, 15);
    v16h b = __builtin_shufflevector(blo, bhi, 0, 1, 2, 3, 4, 5, 6, 7,
                                     8, 9, 10, 11, 12, 13, 14, 15);
    acc = __builtin_amdgcn_wmma_f32_16x16x32_f16(false, a, false, b,
                                                 (short)0, acc, false, false);
    __syncthreads();
  }

  const int n = n0 + lrow;
  if (n < N) {
    float bv = bias ? bias[n] : 0.f;
#pragma unroll
    for (int r = 0; r < 8; ++r) {
      int m = m0 + 16 * wave + r + (half_ ? 8 : 0);
      if (m < M) {
        float val = acc[r] + bv;
        if (relu) val = val > 0.f ? val : 0.f;
        if (residual) val += residual[(size_t)m * N + n];
        Y[(size_t)m * N + n] = val;
      }
    }
  }
}

// Naive linear for tiny K (6->32, 3->C).
__global__ void linear_naive_kernel(const float* __restrict__ X,
                                    const float* __restrict__ W,
                                    const float* __restrict__ bias,
                                    const float* __restrict__ residual,
                                    float* __restrict__ Y,
                                    int M, int K, int N, int relu) {
  size_t t = (size_t)blockIdx.x * blockDim.x + threadIdx.x;
  if (t >= (size_t)M * N) return;
  int m = (int)(t / N), n = (int)(t % N);
  float acc = bias ? bias[n] : 0.f;
  for (int k = 0; k < K; ++k) acc += X[(size_t)m * K + k] * W[(size_t)k * N + n];
  if (relu) acc = acc > 0.f ? acc : 0.f;
  if (residual) acc += residual[t];
  Y[t] = acc;
}

// ---------------------------------------------------------------------------
// KNN: fp32 WMMA Gram tiles + per-row top-16 insertion sort.
// ---------------------------------------------------------------------------
#define KNN_CHUNK 512
__global__ void knn_kernel(const float* __restrict__ P,
                           const float* __restrict__ sq,
                           int* __restrict__ idx, int N, int kOut) {
  __shared__ float dch[16][KNN_CHUNK];
  const int b  = blockIdx.y;
  const int i0 = blockIdx.x * 16;
  const float* Pb  = P  + (size_t)b * N * 3;
  const float* sqb = sq + (size_t)b * N;
  const int tid = threadIdx.x, wave = tid >> 5, lane = tid & 31;
  const int half_ = lane >> 4, lrow = lane & 15;

  const int  ar  = i0 + lrow;
  const bool aok = ar < N;
  v2f afrag;
  afrag.x = aok ? Pb[ar * 3 + (half_ ? 2 : 0)] : 0.f;  // K=0 or K=2
  afrag.y = (aok && !half_) ? Pb[ar * 3 + 1] : 0.f;    // K=1 or K=3(pad)

  float best_d[16];
  int   best_i[16];
#pragma unroll
  for (int q = 0; q < 16; ++q) { best_d[q] = 3.0e38f; best_i[q] = 0; }

  for (int c0 = 0; c0 < N; c0 += KNN_CHUNK) {
    for (int t = wave; t < KNN_CHUNK / 16; t += 8) {
      const int  j0  = c0 + t * 16;
      const int  jc  = j0 + lrow;
      const bool bok = jc < N;
      v2f bfrag;
      bfrag.x = bok ? Pb[jc * 3 + (half_ ? 1 : 0)] : 0.f;  // K=0 / K=1
      bfrag.y = (bok && !half_) ? Pb[jc * 3 + 2] : 0.f;    // K=2 / K=3(pad)
      v8f g = {};
      g = __builtin_amdgcn_wmma_f32_16x16x4_f32(false, afrag, false, bfrag,
                                                (short)0, g, false, false);
      const float sj = bok ? sqb[jc] : 0.f;
#pragma unroll
      for (int r = 0; r < 8; ++r) {
        int mr = r + (half_ ? 8 : 0);
        int gi = i0 + mr;
        float d = (gi < N && bok) ? (sqb[gi] + sj - 2.f * g[r]) : 3.0e38f;
        dch[mr][t * 16 + lrow] = d;
      }
    }
    __syncthreads();
    if (tid < 16) {
      int lim = N - c0; if (lim > KNN_CHUNK) lim = KNN_CHUNK;
      for (int j = 0; j < lim; ++j) {
        float d = dch[tid][j];
        if (d < best_d[15]) {
          float cd = d; int ci = c0 + j;
#pragma unroll
          for (int q = 0; q < 16; ++q) {
            if (cd < best_d[q]) {
              float td = best_d[q]; int ti = best_i[q];
              best_d[q] = cd; best_i[q] = ci;
              cd = td; ci = ti;
            }
          }
        }
      }
    }
    __syncthreads();
  }
  if (tid < 16) {
    int gi = i0 + tid;
    if (gi < N)
      for (int q = 0; q < kOut; ++q)
        idx[((size_t)b * N + gi) * kOut + q] = best_i[q];
  }
}

__global__ void sq_kernel(const float* __restrict__ P, float* __restrict__ sq, int total) {
  int i = blockIdx.x * blockDim.x + threadIdx.x;
  if (i >= total) return;
  float x = P[3 * i], y = P[3 * i + 1], z = P[3 * i + 2];
  sq[i] = x * x + y * y + z * z;
}

// ---------------------------------------------------------------------------
// Elementwise / gather / reduce kernels
// ---------------------------------------------------------------------------
__global__ void concat6_kernel(const float* __restrict__ pos, const float* __restrict__ nrm,
                               float* __restrict__ out, int total) {
  int i = blockIdx.x * blockDim.x + threadIdx.x;
  if (i >= total) return;
#pragma unroll
  for (int c = 0; c < 3; ++c) {
    out[(size_t)i * 6 + c]     = pos[(size_t)i * 3 + c];
    out[(size_t)i * 6 + 3 + c] = nrm[(size_t)i * 3 + c];
  }
}

__global__ void gather_kernel(const float* __restrict__ X, const int* __restrict__ idx,
                              float* __restrict__ out, int Bn, int Nq, int K, int C,
                              int Nsrc, int rowStride) {
  size_t t = (size_t)blockIdx.x * blockDim.x + threadIdx.x;
  size_t total = (size_t)Bn * Nq * K * C;
  if (t >= total) return;
  int c = (int)(t % C); size_t r = t / C;
  int k = (int)(r % K); size_t r2 = r / K;
  int n = (int)(r2 % Nq); int b = (int)(r2 / Nq);
  int src = idx[((size_t)b * Nsrc + (size_t)n * rowStride) * K + k];
  out[t] = X[((size_t)b * Nsrc + src) * C + c];
}

__global__ void rel_kernel(const float* __restrict__ P, const int* __restrict__ idx,
                           float* __restrict__ rel, int Bn, int N, int K) {
  size_t t = (size_t)blockIdx.x * blockDim.x + threadIdx.x;
  size_t total = (size_t)Bn * N * K;
  if (t >= total) return;
  int k = (int)(t % K); size_t r = t / K;
  int n = (int)(r % N); int b = (int)(r / N);
  int src = idx[((size_t)b * N + n) * K + k];
  const float* pb = P + (size_t)b * N * 3;
  float* o = rel + t * 3;
#pragma unroll
  for (int d = 0; d < 3; ++d) o[d] = pb[n * 3 + d] - pb[src * 3 + d];
}

__global__ void apre_kernel(const float* __restrict__ q, float* __restrict__ knb,
                            const float* __restrict__ pe, int Bn, int N, int K, int C) {
  size_t t = (size_t)blockIdx.x * blockDim.x + threadIdx.x;
  size_t total = (size_t)Bn * N * K * C;
  if (t >= total) return;
  int c = (int)(t % C); size_t r = t / C;
  size_t r2 = r / K;
  int n = (int)(r2 % N); int b = (int)(r2 / N);
  knb[t] = q[((size_t)b * N + n) * C + c] - knb[t] + pe[t];
}

__global__ void attn_kernel(const float* __restrict__ a, const float* __restrict__ vnb,
                            const float* __restrict__ pe, float* __restrict__ out,
                            int Bn, int N, int K, int C) {
  size_t t = (size_t)blockIdx.x * blockDim.x + threadIdx.x;
  size_t total = (size_t)Bn * N * C;
  if (t >= total) return;
  int c = (int)(t % C); size_t r = t / C;
  int n = (int)(r % N); int b = (int)(r / N);
  size_t base = (((size_t)b * N + n) * (size_t)K) * C + c;
  float mx = -3.0e38f;
  for (int k = 0; k < K; ++k) mx = fmaxf(mx, a[base + (size_t)k * C]);
  float s = 0.f;
  for (int k = 0; k < K; ++k) s += __expf(a[base + (size_t)k * C] - mx);
  float inv = 1.f / s;
  float acc = 0.f;
  for (int k = 0; k < K; ++k) {
    size_t o = base + (size_t)k * C;
    acc += __expf(a[o] - mx) * inv * (vnb[o] + pe[o]);
  }
  out[((size_t)b * N + n) * C + c] = acc;
}

__global__ void bn_stats_kernel(const float* __restrict__ X, float* __restrict__ stats,
                                int M, int C) {
  __shared__ float s1[256], s2[256];
  int c = blockIdx.x;
  float a = 0.f, b = 0.f;
  for (int m = threadIdx.x; m < M; m += 256) {
    float v = X[(size_t)m * C + c];
    a += v; b += v * v;
  }
  s1[threadIdx.x] = a; s2[threadIdx.x] = b;
  __syncthreads();
  for (int st = 128; st > 0; st >>= 1) {
    if ((int)threadIdx.x < st) {
      s1[threadIdx.x] += s1[threadIdx.x + st];
      s2[threadIdx.x] += s2[threadIdx.x + st];
    }
    __syncthreads();
  }
  if (threadIdx.x == 0) {
    float mu = s1[0] / M;
    stats[c] = mu;
    stats[C + c] = s2[0] / M - mu * mu;
  }
}

__global__ void bn_apply_kernel(float* __restrict__ X, const float* __restrict__ stats,
                                const float* __restrict__ g, const float* __restrict__ bt,
                                int M, int C, int leaky) {
  size_t t = (size_t)blockIdx.x * blockDim.x + threadIdx.x;
  if (t >= (size_t)M * C) return;
  int c = (int)(t % C);
  float v = (X[t] - stats[c]) * rsqrtf(stats[C + c] + BN_EPS) * g[c] + bt[c];
  if (leaky) v = v > 0.f ? v : 0.2f * v;
  X[t] = v;
}

__global__ void maxk_kernel(const float* __restrict__ X, float* __restrict__ out,
                            int Bn, int N, int K, int C) {
  size_t t = (size_t)blockIdx.x * blockDim.x + threadIdx.x;
  size_t total = (size_t)Bn * N * C;
  if (t >= total) return;
  int c = (int)(t % C); size_t r = t / C;
  int n = (int)(r % N); int b = (int)(r / N);
  size_t base = (((size_t)b * N + n) * (size_t)K) * C + c;
  float m = X[base];
  for (int k = 1; k < K; ++k) m = fmaxf(m, X[base + (size_t)k * C]);
  out[t] = m;
}

__global__ void downsample_p_kernel(const float* __restrict__ P, float* __restrict__ Po,
                                    int Bn, int Nout, int stride) {
  size_t t = (size_t)blockIdx.x * blockDim.x + threadIdx.x;
  size_t total = (size_t)Bn * Nout * 3;
  if (t >= total) return;
  int d = (int)(t % 3); size_t r = t / 3;
  int n = (int)(r % Nout); int b = (int)(r / Nout);
  int Nin = Nout * stride;
  Po[t] = P[((size_t)b * Nin + (size_t)n * stride) * 3 + d];
}

__global__ void mean_n_kernel(const float* __restrict__ X, float* __restrict__ out,
                              int Bn, int N, int C) {
  int t = blockIdx.x * blockDim.x + threadIdx.x;
  if (t >= Bn * C) return;
  int c = t % C, b = t / C;
  float s = 0.f;
  for (int n = 0; n < N; ++n) s += X[((size_t)b * N + n) * C + c];
  out[t] = s / N;
}

// ---------------------------------------------------------------------------
// Host-side parameter structs (jax pytree: dict keys alphabetical)
// ---------------------------------------------------------------------------
struct LinP { const float* b; const float* w; };
struct BnP  { const float* bt; const float* g; };
struct PtbP { LinP a1, a2, k, lin1, lin2, p1, p2, q, v; };
struct TdP  { BnP bn; LinP lin; };

static void launch_linear(const float* X, const LinP& l, const float* residual,
                          float* Y, int M, int K, int N, int relu, hipStream_t s) {
  if ((K & 31) == 0) {
    dim3 g(ceildiv(N, 16), ceildiv(M, 64));
    linear_wmma_kernel<<<g, 128, 0, s>>>(X, l.w, l.b, residual, Y, M, K, N, relu);
  } else {
    size_t tot = (size_t)M * N;
    linear_naive_kernel<<<(int)((tot + 255) / 256), 256, 0, s>>>(X, l.w, l.b, residual,
                                                                 Y, M, K, N, relu);
  }
}

static void launch_bn(float* X, const BnP& bn, float* stats, int M, int C,
                      int leaky, hipStream_t s) {
  bn_stats_kernel<<<C, 256, 0, s>>>(X, stats, M, C);
  size_t tot = (size_t)M * C;
  bn_apply_kernel<<<(int)((tot + 255) / 256), 256, 0, s>>>(X, stats, bn.g, bn.bt, M, C, leaky);
}

// ---------------------------------------------------------------------------
extern "C" void kernel_launch(void* const* d_in, const int* in_sizes, int n_in,
                              void* d_out, int out_size, void* d_ws, size_t ws_size,
                              hipStream_t stream) {
  const float* pos = (const float*)d_in[0];
  const float* nrm = (const float*)d_in[1];

  int cur = 2;
  auto nextp   = [&]() { return (const float*)d_in[cur++]; };
  auto readLin = [&](LinP& l) { l.b = nextp(); l.w = nextp(); };
  auto readBn  = [&](BnP& b)  { b.bt = nextp(); b.g = nextp(); };
  auto readPtb = [&](PtbP& p) {
    readLin(p.a1); readLin(p.a2); readLin(p.k); readLin(p.lin1); readLin(p.lin2);
    readLin(p.p1); readLin(p.p2); readLin(p.q); readLin(p.v);
  };

  // params keys sorted: fc, mlp1, ptbs1..5, td1..4
  BnP fbn1, fbn2; LinP fl1, fl2, fl3;
  readBn(fbn1); readBn(fbn2); readLin(fl1); readLin(fl2); readLin(fl3);
  BnP mbn1, mbn2; LinP ml1, ml2;
  readBn(mbn1); readBn(mbn2); readLin(ml1); readLin(ml2);
  PtbP ptbs[13];
  for (int i = 0; i < 13; ++i) readPtb(ptbs[i]);
  TdP tds[4];
  for (int i = 0; i < 4; ++i) { readBn(tds[i].bn); readLin(tds[i].lin); }

  char* wsp = (char*)d_ws;
  size_t off = 0;
  auto alloc = [&](size_t bytes) -> void* {
    off = (off + 255) & ~(size_t)255;
    void* p = wsp + off;
    off += bytes;
    return p;
  };

  const int Ns[5] = {2048, 512, 128, 32, 8};
  const int Cs[5] = {32, 64, 128, 256, 512};
  const int Ks[5] = {16, 16, 16, 16, 8};
  const int nPtb[5] = {1, 2, 3, 5, 2};

  float* stats = (float*)alloc(2 * 512 * sizeof(float));
  float* xA = (float*)alloc((size_t)BATCH * 2048 * 32 * sizeof(float));
  float* xB = (float*)alloc((size_t)BATCH * 2048 * 32 * sizeof(float));
  float* pA = (float*)alloc((size_t)BATCH * 512 * 3 * sizeof(float));
  float* pB = (float*)alloc((size_t)BATCH * 512 * 3 * sizeof(float));
  int*   idxBuf = (int*)alloc((size_t)BATCH * 2048 * 16 * sizeof(int));
  float* sqBuf  = (float*)alloc((size_t)BATCH * 2048 * sizeof(float));

  // --- stem MLP: concat(pos,norm) -> 32 -> 32 ---
  {
    int M = BATCH * 2048;
    size_t ck = off;
    float* cbuf = (float*)alloc((size_t)M * 6 * sizeof(float));
    concat6_kernel<<<ceildiv(M, 256), 256, 0, stream>>>(pos, nrm, cbuf, M);
    launch_linear(cbuf, ml1, nullptr, xA, M, 6, 32, 0, stream);
    launch_bn(xA, mbn1, stats, M, 32, 1, stream);
    launch_linear(xA, ml2, nullptr, xB, M, 32, 32, 0, stream);
    launch_bn(xB, mbn2, stats, M, 32, 1, stream);
    off = ck;
  }
  float* x = xB, *xalt = xA;
  const float* p = pos;
  float* pNextBufs[2] = {pA, pB};
  int pSel = 0, ptbCursor = 0;

  for (int lev = 0; lev < 5; ++lev) {
    const int N = Ns[lev], C = Cs[lev], K = Ks[lev], M = BATCH * N;
    const size_t MK = (size_t)M * K;
    size_t ck = off;

    float* bufH = (float*)alloc((size_t)M * C * sizeof(float));
    float* bufQ = (float*)alloc((size_t)M * C * sizeof(float));
    float* bufK = (float*)alloc((size_t)M * C * sizeof(float));
    float* bufV = (float*)alloc((size_t)M * C * sizeof(float));
    float* nb1  = (float*)alloc(MK * C * sizeof(float));
    float* nb2  = (float*)alloc(MK * C * sizeof(float));
    float* nb3  = (float*)alloc(MK * C * sizeof(float));
    float* nb4  = (float*)alloc(MK * C * sizeof(float));
    float* nb5  = (float*)alloc(MK * C * sizeof(float));
    float* relB = (float*)alloc(MK * 3 * sizeof(float));

    sq_kernel<<<ceildiv(BATCH * N, 256), 256, 0, stream>>>(p, sqBuf, BATCH * N);
    {
      dim3 g(ceildiv(N, 16), BATCH);
      knn_kernel<<<g, 256, 0, stream>>>(p, sqBuf, idxBuf, N, K);
    }

    for (int j = 0; j < nPtb[lev]; ++j) {
      const PtbP& pr = ptbs[ptbCursor++];
      launch_linear(x, pr.lin1, nullptr, bufH, M, C, C, 0, stream);
      launch_linear(bufH, pr.q, nullptr, bufQ, M, C, C, 0, stream);
      launch_linear(bufH, pr.k, nullptr, bufK, M, C, C, 0, stream);
      launch_linear(bufH, pr.v, nullptr, bufV, M, C, C, 0, stream);
      size_t totNb = MK * C;
      gather_kernel<<<(int)((totNb + 255) / 256), 256, 0, stream>>>(
          bufK, idxBuf, nb1, BATCH, N, K, C, N, 1);
      gather_kernel<<<(int)((totNb + 255) / 256), 256, 0, stream>>>(
          bufV, idxBuf, nb2, BATCH, N, K, C, N, 1);
      rel_kernel<<<(int)((MK + 255) / 256), 256, 0, stream>>>(p, idxBuf, relB, BATCH, N, K);
      launch_linear(relB, pr.p1, nullptr, nb3, (int)MK, 3, C, 1, stream);
      launch_linear(nb3, pr.p2, nullptr, nb4, (int)MK, C, C, 0, stream);
      apre_kernel<<<(int)((totNb + 255) / 256), 256, 0, stream>>>(
          bufQ, nb1, nb4, BATCH, N, K, C);
      launch_linear(nb1, pr.a1, nullptr, nb3, (int)MK, C, C, 1, stream);
      launch_linear(nb3, pr.a2, nullptr, nb5, (int)MK, C, C, 0, stream);
      attn_kernel<<<(int)(((size_t)M * C + 255) / 256), 256, 0, stream>>>(
          nb5, nb2, nb4, bufH, BATCH, N, K, C);
      launch_linear(bufH, pr.lin2, x, xalt, M, C, C, 0, stream);
      float* tmp = x; x = xalt; xalt = tmp;
    }

    if (lev < 4) {
      const TdP& td = tds[lev];
      const int N2 = N / 4, C2 = 2 * C;
      const int M2 = BATCH * N2 * 16;
      size_t totG = (size_t)M2 * C;
      gather_kernel<<<(int)((totG + 255) / 256), 256, 0, stream>>>(
          x, idxBuf, nb1, BATCH, N2, 16, C, N, 4);
      launch_linear(nb1, td.lin, nullptr, nb2, M2, C, C2, 0, stream);
      launch_bn(nb2, td.bn, stats, M2, C2, 1, stream);
      size_t totO = (size_t)BATCH * N2 * C2;
      maxk_kernel<<<(int)((totO + 255) / 256), 256, 0, stream>>>(
          nb2, xalt, BATCH, N2, 16, C2);
      float* tmp = x; x = xalt; xalt = tmp;
      float* pn = pNextBufs[pSel]; pSel ^= 1;
      size_t totP = (size_t)BATCH * N2 * 3;
      downsample_p_kernel<<<(int)((totP + 255) / 256), 256, 0, stream>>>(
          p, pn, BATCH, N2, 4);
      p = pn;
    }
    off = ck;
  }

  float* meanB = (float*)alloc((size_t)BATCH * 512 * sizeof(float));
  float* f1    = (float*)alloc((size_t)BATCH * 256 * sizeof(float));
  float* f2    = (float*)alloc((size_t)BATCH * 128 * sizeof(float));
  mean_n_kernel<<<ceildiv(BATCH * 512, 256), 256, 0, stream>>>(x, meanB, BATCH, 8, 512);
  launch_linear(meanB, fl1, nullptr, f1, BATCH, 512, 256, 0, stream);
  launch_bn(f1, fbn1, stats, BATCH, 256, 1, stream);
  launch_linear(f1, fl2, nullptr, f2, BATCH, 256, 128, 0, stream);
  launch_bn(f2, fbn2, stats, BATCH, 128, 1, stream);
  launch_linear(f2, fl3, nullptr, (float*)d_out, BATCH, 128, 40, 0, stream);
}
